// PhaseAttention_16415365005314
// MI455X (gfx1250) — matrix operand order repaired and checked by reading, not compiled, not run on hardware
//
#include <hip/hip_runtime.h>
#include <math.h>

// ---------------------------------------------------------------------------
// PhaseAttention for MI455X (gfx1250, wave32, WMMA + Tensor Data Mover).
//
// Pipeline (bf16 WMMA, fp32 accumulate):
//   1) pack_z:   z fp32 [B,L,D,2]   -> Xcat bf16 [8192][2048] = [Re | Im]
//      pack_w:   W*_r/W*_i fp32     -> Wcat_r=[Wr|-Wi], Wcat_i=[Wi|Wr] bf16
//   2) gemm_dual: C_r,C_i [8192][1024] = Xcat @ Wcat^T.  A/B tiles staged by
//      TENSOR_LOAD_TO_LDS (TDM), double-buffered on TENSORcnt, WMMA compute.
//   3) attn:     per (b,h,chunk) banded attention, online softmax; K tiles
//      staged by TDM, V transposed by VALU; scores & PV via WMMA.
//   4) gemm_dual<F32OUT>: final complex projection, fp32 interleaved out.
//
// Workspace (bf16 halves), ~202 MB:
//   Xcat 16.7M | 8x Wcat 2.1M | 6x QKV planes 8.4M | Ocat 16.7M
// ---------------------------------------------------------------------------

typedef unsigned short     u16;
typedef unsigned int       u32;
typedef unsigned long long u64;
typedef __attribute__((ext_vector_type(16))) __bf16 bf16x16;
typedef __attribute__((ext_vector_type(8)))  float  f32x8;
typedef __attribute__((ext_vector_type(4)))  unsigned int v4u;
typedef __attribute__((ext_vector_type(8)))  int v8i;
typedef __attribute__((ext_vector_type(4)))  int v4i;

struct alignas(16) U4 { u32 x, y, z, w; };
struct alignas(32) Frag { U4 lo, hi; };   // 16 bf16 = one WMMA A/B operand

#define BATCH   2
#define LSEQ    4096
#define DMODEL  1024
#define NHEAD   16
#define HDIM    64
#define CBLK    256
#define MTOK    (BATCH * LSEQ)   // 8192
#define KCAT    (2 * DMODEL)     // 2048

__device__ __forceinline__ U4 ldg4(const u16* p) {
  return *reinterpret_cast<const U4*>(p);
}
__device__ __forceinline__ bf16x16 frag_cast(const Frag& f) {
  return __builtin_bit_cast(bf16x16, f);
}
__device__ __forceinline__ f32x8 wmma_bf16(const Frag& a, const Frag& b, f32x8 c) {
  return __builtin_amdgcn_wmma_f32_16x16x32_bf16(
      /*neg_a=*/false, frag_cast(a), /*neg_b=*/false, frag_cast(b),
      /*c_mod=*/(short)0, c, /*reuse_a=*/false, /*reuse_b=*/false);
}
__device__ __forceinline__ u16 f2bf(float f) {   // round-to-nearest-even
  u32 u = __float_as_uint(f);
  u32 r = u + 0x7FFFu + ((u >> 16) & 1u);
  return (u16)(r >> 16);
}
__device__ __forceinline__ float rmax16(float x) {  // reduce across 16-lane group
  #pragma unroll
  for (int m = 1; m <= 8; m <<= 1) x = fmaxf(x, __shfl_xor(x, m, 32));
  return x;
}
__device__ __forceinline__ float rsum16(float x) {
  #pragma unroll
  for (int m = 1; m <= 8; m <<= 1) x += __shfl_xor(x, m, 32);
  return x;
}

// ---------------------------------------------------------------------------
// Tensor Data Mover: 2D tile (tile_h rows x tile_w elems, 2-byte elements)
// from global (row pitch row_stride_elems) into contiguous LDS at lds_byte.
// D# bit layout per CDNA5 ISA ch.8 (group0: count/lds/global/type,
// group1: data_size, dims, tile dims, dim0 stride). Groups 2/3 zero (2D).
// ---------------------------------------------------------------------------
__device__ __forceinline__ void tdm_load_2d(const void* gaddr, u32 lds_byte,
                                            u32 tile_w, u32 tile_h,
                                            u32 row_stride_elems) {
  const u64 ga = (u64)(size_t)gaddr;
  v4u g0;
  g0[0] = 1u;                                          // count=1 (valid, user)
  g0[1] = lds_byte;                                    // lds_addr
  g0[2] = (u32)ga;                                     // global_addr[31:0]
  g0[3] = (u32)((ga >> 32) & 0x01FFFFFFu) | (2u << 30);// addr[56:32] | type=2
  const u32 dim0 = 1u << 20, dim1 = 1u << 20;          // large (no OOB clip)
  u32 w[8];
  w[0] = 1u << 16;                                     // data_size=1 -> 2B
  w[1] = (dim0 & 0xFFFFu) << 16;                       // tensor_dim0[15:0]
  w[2] = (dim0 >> 16) | ((dim1 & 0xFFFFu) << 16);      // dim0[31:16]|dim1[15:0]
  w[3] = (dim1 >> 16) | (tile_w << 16);                // dim1[31:16]|tile_dim0
  w[4] = tile_h;                                       // tile_dim1 (tile_dim2=0)
  w[5] = row_stride_elems;                             // tensor_dim0_stride
  w[6] = 0u; w[7] = 0u;
  v8i g1;
  #pragma unroll
  for (int i = 0; i < 8; i++) g1[i] = (int)w[i];
  const v4i gz = (v4i)0;
#if __clang_major__ >= 23
  const v8i gz8 = (v8i)0;
  __builtin_amdgcn_tensor_load_to_lds(g0, g1, gz, gz, gz8, 0);
#else
  __builtin_amdgcn_tensor_load_to_lds(g0, g1, gz, gz, 0);
#endif
}

// ---------------------------------------------------------------------------
// Pack kernels (pure bandwidth; negligible vs GEMMs)
// ---------------------------------------------------------------------------
__global__ __launch_bounds__(256)
void pack_z_kernel(const float* __restrict__ z, u16* __restrict__ Xcat) {
  const size_t idx = (size_t)blockIdx.x * 256 + threadIdx.x;  // MTOK*DMODEL
  const size_t t = idx >> 10, d = idx & 1023;
  const float re = z[(t * DMODEL + d) * 2 + 0];
  const float im = z[(t * DMODEL + d) * 2 + 1];
  Xcat[t * KCAT + d]          = f2bf(re);
  Xcat[t * KCAT + DMODEL + d] = f2bf(im);
}

__global__ __launch_bounds__(256)
void pack_w_kernel(const float* __restrict__ Wr, const float* __restrict__ Wi,
                   u16* __restrict__ WcatR, u16* __restrict__ WcatI) {
  const size_t idx = (size_t)blockIdx.x * 256 + threadIdx.x;  // 1024*1024
  const size_t n = idx >> 10, k = idx & 1023;
  const float wr = Wr[n * DMODEL + k];
  const float wi = Wi[n * DMODEL + k];
  WcatR[n * KCAT + k]          = f2bf(wr);    // yr = xr@Wr^T - xi@Wi^T
  WcatR[n * KCAT + DMODEL + k] = f2bf(-wi);
  WcatI[n * KCAT + k]          = f2bf(wi);    // yi = xr@Wi^T + xi@Wr^T
  WcatI[n * KCAT + DMODEL + k] = f2bf(wr);
}

// ---------------------------------------------------------------------------
// Dual-plane GEMM: C_p[m][n] = sum_k A[m][k] * Bp[n][k],  p in {0,1}
// 128x128 tile / block, 16 waves (4x4). TDM-staged, double-buffered LDS:
//   bufA[2]:   2 x [128][64]   (32 KB)
//   bufB[2][2]: 4 x [128][64]  (64 KB)   -> 96 KB dynamic LDS
// ---------------------------------------------------------------------------
template <bool F32OUT>
__global__ __launch_bounds__(512)
void gemm_dual(const u16* __restrict__ A,   // [MTOK][KCAT]
               const u16* __restrict__ B0,  // [1024][KCAT] (B col n = row n)
               const u16* __restrict__ B1,
               u16* __restrict__ C0, u16* __restrict__ C1,
               float* __restrict__ outc) {  // [MTOK][1024][2] if F32OUT
  extern __shared__ __align__(16) u16 gsm[];
  const int tid = threadIdx.x;
  const int lane = tid & 31, wid = tid >> 5;
  const int waveM = wid >> 2, waveN = wid & 3;
  const int mBase = blockIdx.y * 128, nBase = blockIdx.x * 128;
  const int lo = lane & 15, hi = lane >> 4;
  const u32 ldsBase = (u32)(size_t)gsm;   // LDS byte offset (addr[31:0])

  f32x8 acc[2][2][2];
  #pragma unroll
  for (int p = 0; p < 2; p++)
    #pragma unroll
    for (int i = 0; i < 2; i++)
      #pragma unroll
      for (int j = 0; j < 2; j++) acc[p][i][j] = {};

  // Issue the three TDM tile loads (A, B0, B1) for K-slice kk into buffer bi.
  auto issue = [&](int kk, int bi) {
    tdm_load_2d(A  + (size_t)mBase * KCAT + kk, ldsBase + (u32)bi * 16384u,
                64, 128, KCAT);
    tdm_load_2d(B0 + (size_t)nBase * KCAT + kk,
                ldsBase + 32768u + (u32)(bi * 2 + 0) * 16384u, 64, 128, KCAT);
    tdm_load_2d(B1 + (size_t)nBase * KCAT + kk,
                ldsBase + 32768u + (u32)(bi * 2 + 1) * 16384u, 64, 128, KCAT);
  };

  if (wid == 0) { issue(0, 0); __builtin_amdgcn_s_wait_tensorcnt(0); }
  __syncthreads();

  for (int it = 0; it < KCAT / 64; ++it) {
    const int bi = it & 1;
    if (wid == 0 && it + 1 < KCAT / 64) issue((it + 1) * 64, bi ^ 1);

    const u16* sA  = gsm + bi * 8192;
    const u16* sBp[2] = { gsm + 16384 + (bi * 2 + 0) * 8192,
                          gsm + 16384 + (bi * 2 + 1) * 8192 };
    #pragma unroll
    for (int kb = 0; kb < 64; kb += 32) {
      // A fragments: lane<16 row=lo K=kb..kb+15; lane>=16 K=kb+16..31
      Frag af[2];
      #pragma unroll
      for (int mt = 0; mt < 2; mt++) {
        const u16* base = sA + (waveM * 32 + mt * 16 + lo) * 64 + kb;
        af[mt].lo = ldg4(base + hi * 16);
        af[mt].hi = ldg4(base + hi * 16 + 8);
      }
      #pragma unroll
      for (int p = 0; p < 2; p++) {
        #pragma unroll
        for (int nt = 0; nt < 2; nt++) {
          // B fragment: lane<16 K=kb..kb+15 of col lo; lane>=16 K=kb+16..31
          Frag bfr;
          const u16* base = sBp[p] + (waveN * 32 + nt * 16 + lo) * 64 + kb + hi * 16;
          bfr.lo = ldg4(base);
          bfr.hi = ldg4(base + 8);
          #pragma unroll
          for (int mt = 0; mt < 2; mt++)
            acc[p][mt][nt] = wmma_bf16(af[mt], bfr, acc[p][mt][nt]);
        }
      }
    }
    if (wid == 0) __builtin_amdgcn_s_wait_tensorcnt(0);  // next buffer landed
    __syncthreads();
  }

  // D layout: vgpr v -> M = v + 8*hi, N = lo (within 16x16 tile)
  #pragma unroll
  for (int p = 0; p < 2; p++)
    #pragma unroll
    for (int mt = 0; mt < 2; mt++)
      #pragma unroll
      for (int nt = 0; nt < 2; nt++) {
        const int n = nBase + waveN * 32 + nt * 16 + lo;
        #pragma unroll
        for (int v = 0; v < 8; v++) {
          const int m = mBase + waveM * 32 + mt * 16 + v + 8 * hi;
          const float val = acc[p][mt][nt][v];
          if constexpr (F32OUT) {
            outc[((size_t)m * DMODEL + n) * 2 + p] = val;
          } else {
            u16* Cp = p ? C1 : C0;
            Cp[(size_t)m * DMODEL + n] = f2bf(val);
          }
        }
      }
}

// ---------------------------------------------------------------------------
// Banded attention, one block per (b,h,chunk). 512 threads = 16 waves,
// wave w owns query rows [16w, 16w+16). K tiles staged by TDM; V staged
// transposed by VALU; online (flash) softmax over 16 key-pairs.
// Dynamic LDS 276 KB: kS 128K | vT 128K | w-scratch 20K.
// ---------------------------------------------------------------------------
__global__ __launch_bounds__(512)
void attn_kernel(const u16* __restrict__ Qr, const u16* __restrict__ Qi,
                 const u16* __restrict__ Kr, const u16* __restrict__ Ki,
                 const u16* __restrict__ Vr, const u16* __restrict__ Vi,
                 u16* __restrict__ Ocat) {
  extern __shared__ __align__(16) u16 asm_lds[];
  u16* kS[2] = { asm_lds, asm_lds + 512 * 64 };                      // [512][64]
  u16* vT[2] = { asm_lds + 2 * 512 * 64, asm_lds + 3 * 512 * 64 };   // [64][512]
  u16* wS    = asm_lds + 4 * 512 * 64;                               // 16 x 640

  const int tid = threadIdx.x, lane = tid & 31, wid = tid >> 5;
  const int lo = lane & 15, hi = lane >> 4;
  const int bid = blockIdx.x;
  const int c = bid & 15, h = (bid >> 4) & 15, b = bid >> 8;
  const size_t rowBase = (size_t)b * LSEQ;

  // ---- TDM: stage K halves [256][64] each (prev clamped for c==0) ----
  if (wid == 0) {
    const size_t prevRow = rowBase + (size_t)(c > 0 ? (c - 1) * CBLK : 0);
    const size_t curRow  = rowBase + (size_t)c * CBLK;
    const u32 base = (u32)(size_t)asm_lds;
    tdm_load_2d(Kr + prevRow * DMODEL + h * HDIM, base,                64, CBLK, DMODEL);
    tdm_load_2d(Kr + curRow  * DMODEL + h * HDIM, base + 32768u,       64, CBLK, DMODEL);
    tdm_load_2d(Ki + prevRow * DMODEL + h * HDIM, base + 65536u,       64, CBLK, DMODEL);
    tdm_load_2d(Ki + curRow  * DMODEL + h * HDIM, base + 98304u,       64, CBLK, DMODEL);
  }

  // ---- all threads: stage V transposed (vT[d][jj]) while TDM runs ----
  {
    const int jj = tid;  // 0..511 over [prev block | cur block]
    const long src = (long)(c - 1) * CBLK + jj;
    const size_t t = rowBase + (src >= 0 ? (size_t)src : 0);  // clamped; masked
    const u16* vp[2] = { Vr + t * DMODEL + h * HDIM, Vi + t * DMODEL + h * HDIM };
    #pragma unroll
    for (int comp = 0; comp < 2; comp++) {
      #pragma unroll
      for (int i = 0; i < 8; i++) {
        const U4 q = ldg4(vp[comp] + i * 8);
        const u32 wd[4] = { q.x, q.y, q.z, q.w };
        #pragma unroll
        for (int e = 0; e < 4; e++) {
          vT[comp][(i * 8 + 2 * e    ) * 512 + jj] = (u16)(wd[e] & 0xFFFFu);
          vT[comp][(i * 8 + 2 * e + 1) * 512 + jj] = (u16)(wd[e] >> 16);
        }
      }
    }
  }
  if (wid == 0) __builtin_amdgcn_s_wait_tensorcnt(0);
  __syncthreads();

  // ---- per-wave Q fragments (A operand), loaded once from global ----
  Frag qf[2][2];  // [comp][kb]
  {
    const size_t t = rowBase + (size_t)c * CBLK + wid * 16 + lo;
    const u16* qp[2] = { Qr + t * DMODEL + h * HDIM, Qi + t * DMODEL + h * HDIM };
    #pragma unroll
    for (int comp = 0; comp < 2; comp++)
      #pragma unroll
      for (int kb = 0; kb < 2; kb++) {
        qf[comp][kb].lo = ldg4(qp[comp] + kb * 32 + hi * 16);
        qf[comp][kb].hi = ldg4(qp[comp] + kb * 32 + hi * 16 + 8);
      }
  }

  f32x8 oacc[2][4];
  #pragma unroll
  for (int comp = 0; comp < 2; comp++)
    #pragma unroll
    for (int dt = 0; dt < 4; dt++) oacc[comp][dt] = {};
  float Mr[8], Sr[8];
  #pragma unroll
  for (int v = 0; v < 8; v++) { Mr[v] = -__builtin_inff(); Sr[v] = 0.f; }

  u16* wbase = wS + wid * 640;          // [16][stride 40] prob-tile scratch
  const int lim = (c > 0) ? 0 : CBLK;   // chunk 0: prev half fully masked

  for (int tp = 0; tp < 16; tp++) {     // 16 key-pairs of 32 keys
    const int jj0 = tp * 32;
    // ---- scores: S = Qr*Kr^T + Qi*Ki^T over HD=64 (2 K-steps) ----
    f32x8 s[2];
    #pragma unroll
    for (int st = 0; st < 2; st++) {
      f32x8 a = {};
      const int jj = jj0 + st * 16 + lo;
      #pragma unroll
      for (int comp = 0; comp < 2; comp++)
        #pragma unroll
        for (int kb = 0; kb < 2; kb++) {
          Frag bfr;
          const u16* base = &kS[comp][jj * 64 + kb * 32 + hi * 16];
          bfr.lo = ldg4(base);
          bfr.hi = ldg4(base + 8);
          a = wmma_bf16(qf[comp][kb], bfr, a);
        }
      s[st] = a;
    }
    // ---- scale + band mask + online softmax update ----
    #pragma unroll
    for (int v = 0; v < 8; v++) {
      const int m = wid * 16 + v + 8 * hi;  // query index within chunk
      float best = -__builtin_inff();
      #pragma unroll
      for (int st = 0; st < 2; st++) {
        const int jj = jj0 + st * 16 + lo;
        float val = s[st][v] * 0.125f;      // HD^-0.5
        // band: jj in (m, m+256]  <=>  (u32)(jj-m-1) < 256; plus chunk-0 clip
        const bool ok = ((u32)(jj - m - 1) < 256u) && (jj >= lim);
        val = ok ? val : -1.0e30f;
        s[st][v] = val;
        best = fmaxf(best, val);
      }
      best = rmax16(best);
      const float newM = fmaxf(Mr[v], best);
      const float alpha = __expf(Mr[v] - newM);
      Mr[v] = newM;
      float psum = 0.f;
      #pragma unroll
      for (int st = 0; st < 2; st++) {
        const float p = __expf(s[st][v] - newM);
        psum += p;
        wbase[(v + 8 * hi) * 40 + st * 16 + lo] = f2bf(p);  // transpose via LDS
      }
      psum = rsum16(psum);
      Sr[v] = Sr[v] * alpha + psum;
      #pragma unroll
      for (int comp = 0; comp < 2; comp++)
        #pragma unroll
        for (int dt = 0; dt < 4; dt++) oacc[comp][dt][v] *= alpha;
    }
    __threadfence_block();  // order per-wave LDS scratch write -> read

    // ---- O += P @ V over 32 keys (P from scratch in A layout) ----
    Frag wf;
    {
      const u16* basew = wbase + lo * 40;
      wf.lo = ldg4(basew + hi * 16);
      wf.hi = ldg4(basew + hi * 16 + 8);
    }
    #pragma unroll
    for (int comp = 0; comp < 2; comp++)
      #pragma unroll
      for (int dt = 0; dt < 4; dt++) {
        Frag bfr;
        const u16* base = &vT[comp][(dt * 16 + lo) * 512 + jj0 + hi * 16];
        bfr.lo = ldg4(base);
        bfr.hi = ldg4(base + 8);
        oacc[comp][dt] = wmma_bf16(wf, bfr, oacc[comp][dt]);
      }
  }

  // ---- normalize and emit Ocat = [Or | Oi] bf16 ----
  #pragma unroll
  for (int comp = 0; comp < 2; comp++)
    #pragma unroll
    for (int dt = 0; dt < 4; dt++)
      #pragma unroll
      for (int v = 0; v < 8; v++) {
        const int m = wid * 16 + v + 8 * hi;
        const int d = dt * 16 + lo;
        const size_t t = rowBase + (size_t)c * CBLK + m;
        const float val = oacc[comp][dt][v] / Sr[v];
        Ocat[t * KCAT + comp * DMODEL + h * HDIM + d] = f2bf(val);
      }
}

// ---------------------------------------------------------------------------
// Host launcher. Workspace need: ~202 MB of bf16 halves.
// ---------------------------------------------------------------------------
extern "C" void kernel_launch(void* const* d_in, const int* in_sizes, int n_in,
                              void* d_out, int out_size, void* d_ws, size_t ws_size,
                              hipStream_t stream) {
  (void)in_sizes; (void)n_in; (void)out_size; (void)ws_size;
  const float* z = (const float*)d_in[0];
  const float* W[8];
  for (int i = 0; i < 8; i++) W[i] = (const float*)d_in[1 + i];

  u16* ws = (u16*)d_ws;
  u16* Xcat = ws;                                             // 16,777,216 halves
  u16* Wcat[8];
  for (int i = 0; i < 8; i++) Wcat[i] = ws + 16777216ull + (size_t)i * 2097152ull;
  u16* plane[6];                                              // Qr Qi Kr Ki Vr Vi
  for (int i = 0; i < 6; i++) plane[i] = ws + 33554432ull + (size_t)i * 8388608ull;
  u16* Ocat = ws + 83886080ull;                               // 16,777,216 halves

  // 1) pack inputs to bf16 / folded-complex layouts
  pack_z_kernel<<<(MTOK * DMODEL) / 256, 256, 0, stream>>>(z, Xcat);
  for (int p = 0; p < 4; p++)
    pack_w_kernel<<<(DMODEL * DMODEL) / 256, 256, 0, stream>>>(
        W[2 * p], W[2 * p + 1], Wcat[2 * p], Wcat[2 * p + 1]);

  // 2) Q/K/V complex projections (TDM double-buffered, 96 KB dynamic LDS)
  dim3 gg(DMODEL / 128, MTOK / 128);  // (8, 64)
  const size_t gemmLds = 98304;
  gemm_dual<false><<<gg, 512, gemmLds, stream>>>(Xcat, Wcat[0], Wcat[1], plane[0], plane[1], nullptr);
  gemm_dual<false><<<gg, 512, gemmLds, stream>>>(Xcat, Wcat[2], Wcat[3], plane[2], plane[3], nullptr);
  gemm_dual<false><<<gg, 512, gemmLds, stream>>>(Xcat, Wcat[4], Wcat[5], plane[4], plane[5], nullptr);

  // 3) banded attention: 276 KB dynamic LDS (<= 320 KB/WGP)
  const size_t attnLds = (size_t)(4 * 512 * 64 + 16 * 640) * sizeof(u16);
  attn_kernel<<<BATCH * NHEAD * (LSEQ / CBLK), 512, attnLds, stream>>>(
      plane[0], plane[1], plane[2], plane[3], plane[4], plane[5], Ocat);

  // 4) output complex projection, fp32 interleaved into d_out
  gemm_dual<true><<<gg, 512, gemmLds, stream>>>(Ocat, Wcat[6], Wcat[7], nullptr, nullptr,
                                                (float*)d_out);
}
